// MultiHeadAttention_19318762897556
// MI455X (gfx1250) — matrix operand order repaired and checked by reading, not compiled, not run on hardware
//
#include <hip/hip_runtime.h>

#define Bq   2
#define Sq   2048
#define DIMq 1024
#define Hq   16
#define DHq  64
#define SPq  1024
#define Lq   3072          // SPq + Sq
#define Mrows 4096         // Bq * Sq

typedef __attribute__((ext_vector_type(16))) __bf16 v16bf;
typedef __attribute__((ext_vector_type(8)))  __bf16 v8bf;
typedef __attribute__((ext_vector_type(8)))  float  v8f;
typedef __attribute__((ext_vector_type(4)))  float  v4f;
typedef __attribute__((ext_vector_type(4)))  int    v4i;

#define AS1 __attribute__((address_space(1)))
#define AS3 __attribute__((address_space(3)))

#if defined(__HIP_DEVICE_COMPILE__) && \
    __has_builtin(__builtin_amdgcn_global_load_async_to_lds_b128) && \
    __has_builtin(__builtin_amdgcn_s_wait_asynccnt)
#define HAVE_ASYNC_LDS 1
#else
#define HAVE_ASYNC_LDS 0
#endif

static __device__ __forceinline__ void async_copy_b128(const void* g, void* l) {
#if HAVE_ASYNC_LDS
    __builtin_amdgcn_global_load_async_to_lds_b128((AS1 v4i*)g, (AS3 v4i*)l, 0, 0);
#endif
}
static __device__ __forceinline__ void async_wait0() {
#if HAVE_ASYNC_LDS
    __builtin_amdgcn_s_wait_asynccnt(0);
#endif
}

static __device__ __forceinline__ v16bf cat8(v8bf lo, v8bf hi) {
    return __builtin_shufflevector(lo, hi, 0,1,2,3,4,5,6,7,8,9,10,11,12,13,14,15);
}
static __device__ __forceinline__ v8f wmma_bf16(v16bf a, v16bf b, v8f c) {
    // D = A(16x32 bf16) * B(32x16 bf16) + C(16x16 f32)
    return __builtin_amdgcn_wmma_f32_16x16x32_bf16(false, a, false, b, (short)0, c, false, false);
}
static __device__ __forceinline__ unsigned pack_bf16(float lo, float hi) {
    unsigned short a = __builtin_bit_cast(unsigned short, (__bf16)lo);
    unsigned short b = __builtin_bit_cast(unsigned short, (__bf16)hi);
    return (unsigned)a | ((unsigned)b << 16);
}

// ---------------------------------------------------------------- elementwise
__global__ void cvt_f32_bf16(const float* __restrict__ src, __bf16* __restrict__ dst, int n) {
    for (int i = blockIdx.x * blockDim.x + threadIdx.x; i < n; i += gridDim.x * blockDim.x)
        dst[i] = (__bf16)src[i];
}

// incr_state [B,H,2,SP,DH] (f32) -> K/V prefix [B,H,L,DH] (bf16)
__global__ void seed_kv(const float* __restrict__ incr, __bf16* __restrict__ Kb,
                        __bf16* __restrict__ Vb) {
    const int n = Bq * Hq * 2 * SPq * DHq;     // 2^22
    for (int i = blockIdx.x * blockDim.x + threadIdx.x; i < n; i += gridDim.x * blockDim.x) {
        int d = i & (DHq - 1);
        int p = (i >> 6) & (SPq - 1);
        int t = (i >> 16) & 1;
        int h = (i >> 17) & (Hq - 1);
        int b = i >> 21;
        size_t dst = ((size_t)(b * Hq + h) * Lq + p) * DHq + d;
        __bf16 v = (__bf16)incr[i];
        if (t == 0) Kb[dst] = v; else Vb[dst] = v;
    }
}

// pre [4096,3072] bf16 -> Q[B,H,S,DH], K/V[B,H,L,DH] (new part at offset SP)
__global__ void scatter_qkv(const __bf16* __restrict__ pre, __bf16* __restrict__ Qb,
                            __bf16* __restrict__ Kb, __bf16* __restrict__ Vb) {
    const int n = Mrows * 3 * DIMq;
    for (int i = blockIdx.x * blockDim.x + threadIdx.x; i < n; i += gridDim.x * blockDim.x) {
        int r = i / (3 * DIMq), j = i - r * (3 * DIMq);
        int t = j / DIMq, rem = j - t * DIMq;
        int h = rem >> 6, d = rem & (DHq - 1);
        int b = r >> 11, s = r & (Sq - 1);
        __bf16 v = pre[i];
        if (t == 0)
            Qb[((size_t)(b * Hq + h) * Sq + s) * DHq + d] = v;
        else if (t == 1)
            Kb[((size_t)(b * Hq + h) * Lq + SPq + s) * DHq + d] = v;
        else
            Vb[((size_t)(b * Hq + h) * Lq + SPq + s) * DHq + d] = v;
    }
}

// ---------------------------------------------------------------- WMMA GEMM
// C[M,N] = A[M,K] * Bw[K,N] + bias ;  block = 128 thr (4 waves), tile 64x64, kstep 32
template <bool OUTF32>
__global__ void gemm_bf16(const __bf16* __restrict__ A, const __bf16* __restrict__ Bw,
                          const float* __restrict__ bias, float* __restrict__ outF,
                          __bf16* __restrict__ outB, int M, int N, int K) {
    __shared__ alignas(64) __bf16 lA[64][32];
    __shared__ alignas(64) __bf16 lBt[64][32];   // transposed: [n][k]

    const int n0 = blockIdx.x * 64, m0 = blockIdx.y * 64;
    const int tid = threadIdx.x, lane = tid & 31, wid = tid >> 5;
    const int half = lane >> 4, l16 = lane & 15;

    v8f acc[4] = {v8f{}, v8f{}, v8f{}, v8f{}};

    for (int k0 = 0; k0 < K; k0 += 32) {
        { // stage A tile 64x32: verbatim copy -> async-to-LDS path
            int r = tid >> 1, c = (tid & 1) * 16;
            const __bf16* g = A + (size_t)(m0 + r) * K + k0 + c;
#if HAVE_ASYNC_LDS
            async_copy_b128(g,     &lA[r][c]);
            async_copy_b128(g + 8, &lA[r][c + 8]);
#else
            const uint32_t* gw = (const uint32_t*)g;
            uint32_t* s = (uint32_t*)&lA[r][c];
            #pragma unroll
            for (int i = 0; i < 8; ++i) s[i] = gw[i];
#endif
            if (k0 + 32 < K) __builtin_prefetch(g + 32, 0, 0);  // next k-step
        }
        { // stage B tile transposed: lBt[n][k] = Bw[k0+k][n0+n]
            int n = tid & 63, kh = (tid >> 6) * 16;
            #pragma unroll
            for (int i = 0; i < 16; ++i)
                lBt[n][kh + i] = Bw[(size_t)(k0 + kh + i) * N + n0 + n];
        }
        async_wait0();
        __syncthreads();

        const int row = wid * 16 + l16;                // A layout: row = lane&15
        v8bf alo = *(const v8bf*)&lA[row][16 * half];  // K = 16*half .. +15
        v8bf ahi = *(const v8bf*)&lA[row][16 * half + 8];
        v16bf a = cat8(alo, ahi);
        #pragma unroll
        for (int nt = 0; nt < 4; ++nt) {
            // B layout: col = lane&15, K = 16*half + e  -> contiguous in lBt row
            v16bf b = *(const v16bf*)&lBt[nt * 16 + l16][16 * half];
            acc[nt] = wmma_bf16(a, b, acc[nt]);
        }
        __syncthreads();
    }

    const int rbase = m0 + wid * 16 + 8 * half;        // D: row = v + 8*half
    #pragma unroll
    for (int nt = 0; nt < 4; ++nt) {
        int col = n0 + nt * 16 + l16;                  // D: col = lane&15
        float bv = bias[col];
        #pragma unroll
        for (int v = 0; v < 8; ++v) {
            float val = acc[nt][v] + bv;
            size_t idx = (size_t)(rbase + v) * N + col;
            if (OUTF32) outF[idx] = val;
            else        outB[idx] = (__bf16)val;
        }
    }
}

// ---------------------------------------------------------------- attention
// block = 4 waves; each wave: 16 query rows; block: 64 queries of one (b,h).
// Flash-style online softmax over key tiles of 32. Scores computed TRANSPOSED
// (A = K tile -> keys on M, B = Q -> queries on lanes) so the key reduction is
// in-lane VALU + one xor-16 shuffle instead of a 4-step bpermute chain per row.
__global__ void attn_kernel(const __bf16* __restrict__ Qb, const __bf16* __restrict__ Kb,
                            const __bf16* __restrict__ Vb, const float* __restrict__ mask,
                            __bf16* __restrict__ ctxb) {
    __shared__ alignas(64) __bf16 lK[32][64];       // [key][dh]   (natural)
    __shared__ alignas(64) __bf16 lVt[64][32];      // [dh][key]   (transposed)
    __shared__ alignas(64) __bf16 lP[4][16][32];    // per-wave probs [query][key]

    const int qblk = blockIdx.x & 31;               // S/64 = 32
    const int bh = blockIdx.x >> 5;
    const int b = bh >> 4, h = bh & 15;
    const int tid = threadIdx.x, lane = tid & 31, wid = tid >> 5;
    const int half = lane >> 4, l16 = lane & 15;
    const int q0 = qblk * 64 + wid * 16;

    // Q tile in registers as B operand: lane = query col, K = dh = 16*half + e
    // -> two natural contiguous 32B loads per dh chunk.
    v16bf bq[2];
    {
        const __bf16* qp = Qb + ((size_t)(b * Hq + h) * Sq + q0 + l16) * DHq;
        #pragma unroll
        for (int c = 0; c < 2; ++c)
            bq[c] = *(const v16bf*)(qp + c * 32 + 16 * half);
    }

    // per-lane softmax state for query (lane&15), replicated across halves
    float Mr = -__builtin_inff(), Sr = 0.f;
    v8f acc[4] = {v8f{}, v8f{}, v8f{}, v8f{}};     // ctx: queries on M, dh on lanes

    const size_t kvbase = (size_t)(b * Hq + h) * Lq * DHq;
    const float* mrow = mask + ((size_t)b * Sq + q0 + l16) * (size_t)Lq;

    for (int k0 = 0; k0 < Lq; k0 += 32) {
        { // stage K tile 32x64: verbatim copy -> async-to-LDS path
            int key = tid >> 2, c = (tid & 3) * 16;
            const __bf16* g = Kb + kvbase + (size_t)(k0 + key) * DHq + c;
#if HAVE_ASYNC_LDS
            async_copy_b128(g,     &lK[key][c]);
            async_copy_b128(g + 8, &lK[key][c + 8]);
#else
            const uint32_t* gw = (const uint32_t*)g;
            uint32_t* s = (uint32_t*)&lK[key][c];
            #pragma unroll
            for (int i = 0; i < 8; ++i) s[i] = gw[i];
#endif
        }
        { // stage V tile transposed (transpose -> manual path)
            int key = tid >> 2, c = (tid & 3) * 16;
            const __bf16* g = Vb + kvbase + (size_t)(k0 + key) * DHq + c;
            #pragma unroll
            for (int i = 0; i < 16; ++i) lVt[c + i][key] = g[i];
        }
        async_wait0();
        __syncthreads();

        // scores^T: s0 = keys k0..k0+15 on M, s1 = keys k0+16..+31; queries on lanes
        v8f s0 = v8f{}, s1 = v8f{};
        #pragma unroll
        for (int c = 0; c < 2; ++c) {
            // A operand (keys): row = key = lane&15 (+16 for tile1), two 8-el runs
            v8bf a0lo = *(const v8bf*)&lK[l16][c * 32 + 16 * half];
            v8bf a0hi = *(const v8bf*)&lK[l16][c * 32 + 16 * half + 8];
            v8bf a1lo = *(const v8bf*)&lK[16 + l16][c * 32 + 16 * half];
            v8bf a1hi = *(const v8bf*)&lK[16 + l16][c * 32 + 16 * half + 8];
            s0 = wmma_bf16(cat8(a0lo, a0hi), bq[c], s0);
            s1 = wmma_bf16(cat8(a1lo, a1hi), bq[c], s1);
        }
        // + mask: lane owns query (lane&15); its 8 keys are consecutive -> b128 loads
        {
            v4f ma = *(const v4f*)(mrow + k0 + 8 * half);
            v4f mb = *(const v4f*)(mrow + k0 + 8 * half + 4);
            v4f mc = *(const v4f*)(mrow + k0 + 16 + 8 * half);
            v4f md = *(const v4f*)(mrow + k0 + 16 + 8 * half + 4);
            #pragma unroll
            for (int v = 0; v < 4; ++v) {
                s0[v] += ma[v]; s0[4 + v] += mb[v];
                s1[v] += mc[v]; s1[4 + v] += md[v];
            }
        }

        // online softmax, state per lane (query = lane&15)
        float t = fmaxf(s0[0], s1[0]);
        #pragma unroll
        for (int v = 1; v < 8; ++v) t = fmaxf(t, fmaxf(s0[v], s1[v]));
        t = fmaxf(t, __shfl_xor(t, 16, 32));        // merge complementary key halves
        float mn   = fmaxf(Mr, t);
        float corr = __expf(Mr - mn);
        float p0[8], p1[8], rs = 0.f;
        #pragma unroll
        for (int v = 0; v < 8; ++v) {
            p0[v] = __expf(s0[v] - mn);
            p1[v] = __expf(s1[v] - mn);
            rs += p0[v] + p1[v];
        }
        rs = rs + __shfl_xor(rs, 16, 32);
        Sr = Sr * corr + rs;
        Mr = mn;

        // store P^T -> lds_P[query][key] (packed pairs, consecutive keys)
        {
            const int q = l16;
            #pragma unroll
            for (int v = 0; v < 8; v += 2) {
                *(unsigned*)&lP[wid][q][8 * half + v]      = pack_bf16(p0[v], p0[v + 1]);
                *(unsigned*)&lP[wid][q][16 + 8 * half + v] = pack_bf16(p1[v], p1[v + 1]);
            }
        }
        // rescale ctx acc: rows are queries v + 8*half -> broadcast corr from lane domain
        float cv[8];
        #pragma unroll
        for (int v = 0; v < 8; ++v) cv[v] = __shfl(corr, v + 8 * half, 32);
        #pragma unroll
        for (int nt = 0; nt < 4; ++nt)
            #pragma unroll
            for (int v = 0; v < 8; ++v) acc[nt][v] *= cv[v];

        // P in A layout (row = query = lane&15); same-wave DS ops are in order
        v8bf plo = *(const v8bf*)&lP[wid][l16][16 * half];
        v8bf phi = *(const v8bf*)&lP[wid][l16][16 * half + 8];
        v16bf pa = cat8(plo, phi);
        #pragma unroll
        for (int nt = 0; nt < 4; ++nt) {
            // B operand: col = dh (lane&15 + nt*16), K = key = 16*half + e -> contiguous in lVt
            v16bf bv = *(const v16bf*)&lVt[nt * 16 + l16][16 * half];
            acc[nt] = wmma_bf16(pa, bv, acc[nt]);
        }
        __syncthreads();
    }

    // normalize (broadcast Sr into VGPR-row domain) and write ctx [b, s, h*DH+d]
    float sv[8];
    #pragma unroll
    for (int v = 0; v < 8; ++v) sv[v] = __shfl(Sr, v + 8 * half, 32);
    #pragma unroll
    for (int nt = 0; nt < 4; ++nt) {
        int col = h * DHq + nt * 16 + l16;
        #pragma unroll
        for (int v = 0; v < 8; ++v) {
            int q = q0 + v + 8 * half;
            float val = acc[nt][v] / sv[v];
            ctxb[((size_t)b * Sq + q) * DIMq + col] = (__bf16)val;
        }
    }
}

// ---------------------------------------------------------------- launch
extern "C" void kernel_launch(void* const* d_in, const int* in_sizes, int n_in,
                              void* d_out, int out_size, void* d_ws, size_t ws_size,
                              hipStream_t stream) {
    const float* hs    = (const float*)d_in[0];
    const float* mask  = (const float*)d_in[1];
    const float* incr  = (const float*)d_in[2];
    const float* W_pre = (const float*)d_in[3];
    const float* b_pre = (const float*)d_in[4];
    const float* W_out = (const float*)d_in[5];
    const float* b_out = (const float*)d_in[6];
    float* out = (float*)d_out;

    char* w = (char*)d_ws;
    __bf16* Xb   = (__bf16*)w;  w += (size_t)Mrows * DIMq * 2;            // 8 MB
    __bf16* Wpb  = (__bf16*)w;  w += (size_t)DIMq * 3 * DIMq * 2;         // 6 MB
    __bf16* Wob  = (__bf16*)w;  w += (size_t)DIMq * DIMq * 2;             // 2 MB
    __bf16* preb = (__bf16*)w;  w += (size_t)Mrows * 3 * DIMq * 2;        // 24 MB
    __bf16* Qb   = (__bf16*)w;  w += (size_t)Bq * Hq * Sq * DHq * 2;      // 8 MB
    __bf16* Kb   = (__bf16*)w;  w += (size_t)Bq * Hq * Lq * DHq * 2;      // 12 MB
    __bf16* Vb   = (__bf16*)w;  w += (size_t)Bq * Hq * Lq * DHq * 2;      // 12 MB
    __bf16* ctxb = (__bf16*)w;  w += (size_t)Mrows * DIMq * 2;            // 8 MB

    // 1) precision conversions
    cvt_f32_bf16<<<2048, 256, 0, stream>>>(hs, Xb, Mrows * DIMq);
    cvt_f32_bf16<<<2048, 256, 0, stream>>>(W_pre, Wpb, DIMq * 3 * DIMq);
    cvt_f32_bf16<<<1024, 256, 0, stream>>>(W_out, Wob, DIMq * DIMq);
    seed_kv<<<2048, 256, 0, stream>>>(incr, Kb, Vb);

    // 2) QKV projection: pre = X @ W_pre + b_pre  (4096x3072, K=1024)
    gemm_bf16<false><<<dim3(3 * DIMq / 64, Mrows / 64), 128, 0, stream>>>(
        Xb, Wpb, b_pre, nullptr, preb, Mrows, 3 * DIMq, DIMq);
    scatter_qkv<<<4096, 256, 0, stream>>>(preb, Qb, Kb, Vb);

    // 3) attention
    attn_kernel<<<Bq * Hq * (Sq / 64), 128, 0, stream>>>(Qb, Kb, Vb, mask, ctxb);

    // 4) output projection: out = ctx @ W_out + b_out  (4096x1024, K=1024)
    gemm_bf16<true><<<dim3(DIMq / 64, Mrows / 64), 128, 0, stream>>>(
        ctxb, Wob, b_out, out, nullptr, Mrows, DIMq, DIMq);
}
